// FigCrossEntropy_23802708755037
// MI455X (gfx1250) — compile-verified, matrix operand order checked by reading, not run on hardware
//
#include <hip/hip_runtime.h>
#include <hip/hip_bf16.h>
#include <math.h>

// ---- problem constants (fixed by the reference) ----
#define N_TOK   4096
#define DMODEL  2048
#define VOCAB   65536
#define TM      128     // tokens per block
#define TN      128     // vocab cols per sub-tile
#define KK      32      // K step (bf16 WMMA K)
#define KSTEPS  (DMODEL / KK)
#define SUBS    8       // sub-tiles per block -> 1024 cols per block
#define NCHUNKS 64      // 65536 / 1024 vocab stripes
#define IGNORE_INDEX (-100)
#define ABUF    (TM * 40)   // ushorts per staging buffer (80B padded rows)

typedef __attribute__((ext_vector_type(16))) __bf16 v16bf;
typedef __attribute__((ext_vector_type(8)))  float  v8f;

union FragCast { uint4 q[2]; v16bf v; };

__device__ __forceinline__ unsigned short f32_to_bf16(float f) {
  unsigned int u = __float_as_uint(f);
  unsigned int r = u + 0x7FFFu + ((u >> 16) & 1u);   // round-to-nearest-even
  return (unsigned short)(r >> 16);
}

// ---------------------------------------------------------------------------
// Kernel 0: one-shot f32 -> bf16 conversion (pure-bandwidth prepass).
// ---------------------------------------------------------------------------
__global__ __launch_bounds__(256) void cvt_bf16(const float* __restrict__ src,
                                                unsigned short* __restrict__ dst,
                                                int n4 /* float4 count */) {
  int i = blockIdx.x * 256 + threadIdx.x;
  const int stride = gridDim.x * 256;
  for (; i < n4; i += stride) {
    const float4 v = ((const float4*)src)[i];
    uint2 p;
    p.x = (unsigned)f32_to_bf16(v.x) | ((unsigned)f32_to_bf16(v.y) << 16);
    p.y = (unsigned)f32_to_bf16(v.z) | ((unsigned)f32_to_bf16(v.w) << 16);
    ((uint2*)dst)[i] = p;
  }
}

// ---------------------------------------------------------------------------
// Kernel 1: bf16 GEMM + online softmax.
//   - unroll-1 K loop, XOR ping-pong -> accumulators pinned, no mov/nop churn
//   - v_wmma_f32_16x16x32_bf16, 8 WMMA / wave / K-step, f32 accumulation
// ---------------------------------------------------------------------------
__global__ __launch_bounds__(256) void ce_main(
    const unsigned short* __restrict__ Hb, const unsigned short* __restrict__ Wb,
    const int* __restrict__ targets,
    float* __restrict__ pMax, float* __restrict__ pSum, float* __restrict__ pTgt)
{
  __shared__ unsigned short sA[2 * ABUF];     // H tiles, ping-pong
  __shared__ unsigned short sB[2 * ABUF];     // W tiles, ping-pong
  __shared__ float sLogits[TM * 132];         // 128x128 f32 logits (stride 132)
  __shared__ float sRed[256];
  __shared__ float sMax[TM];
  __shared__ float sSum[TM];

  const int tid      = threadIdx.x;
  const int lane     = tid & 31;
  const int wave     = tid >> 5;
  const int wm       = wave >> 1;      // 0..3 -> 32 token rows each
  const int wn       = wave & 1;       // 0..1 -> 64 cols each
  const int laneHalf = lane >> 4;      // 0/1
  const int ln       = lane & 15;

  const int token0  = blockIdx.x * TM;
  const int colBase = blockIdx.y * (SUBS * TN);

  // staging: each thread copies two 16B chunks per tile (rows r0 and r0+64)
  const int r0 = tid >> 2, q0 = tid & 3;                  // 4 threads / row
  const unsigned short* hRow = Hb + (size_t)(token0 + r0) * DMODEL + q0 * 8;
  const int sOff0 = r0 * 40 + q0 * 8;
  const int sOff1 = (r0 + 64) * 40 + q0 * 8;

  // loop-invariant fragment offsets (ushort units)
  const int kk0   = laneHalf * 8;
  const int aOff0 = (wm * 32 + ln) * 40 + kk0;            // m-tile 0
  const int aOff1 = aOff0 + 16 * 40;                      // m-tile 1
  const int bOff  = (wn * 64 + ln) * 40 + kk0;            // n-tile 0 (+16*40*tn)

  const int myTgt = (tid < TM) ? targets[token0 + tid] : -1;
  if (tid < TM) { sMax[tid] = -INFINITY; sSum[tid] = 0.0f; }
  __syncthreads();

  for (int sub = 0; sub < SUBS; ++sub) {
    const int col0 = colBase + sub * TN;
    const unsigned short* wRow = Wb + (size_t)(col0 + r0) * DMODEL + q0 * 8;
    const unsigned short* hPtr = hRow;
    const unsigned short* wPtr = wRow;

    v8f c[2][4] = {};

    // ---- prologue: stage K-step 0 into buffer 0 ----
    uint4 ha0 = *(const uint4*)(hPtr);
    uint4 ha1 = *(const uint4*)(hPtr + (size_t)64 * DMODEL);
    uint4 wv0 = *(const uint4*)(wPtr);
    uint4 wv1 = *(const uint4*)(wPtr + (size_t)64 * DMODEL);
    *(uint4*)&sA[sOff0] = ha0;  *(uint4*)&sA[sOff1] = ha1;
    *(uint4*)&sB[sOff0] = wv0;  *(uint4*)&sB[sOff1] = wv1;

    int bufOff = 0;
    #pragma unroll 1
    for (int step = 0; step < KSTEPS; ++step) {
      __syncthreads();                       // buf[bufOff] ready; other buf free

      // issue next tile's global loads early (latency hidden behind WMMA)
      if (step + 1 < KSTEPS) {
        hPtr += KK;  wPtr += KK;
        ha0 = *(const uint4*)(hPtr);
        ha1 = *(const uint4*)(hPtr + (size_t)64 * DMODEL);
        wv0 = *(const uint4*)(wPtr);
        wv1 = *(const uint4*)(wPtr + (size_t)64 * DMODEL);
      }

      // ---- fragment loads per ISA 16-bit A/B layout ----
      // lanes 0-15 hold K 0..7 & 16..23, lanes 16-31 hold K 8..15 & 24..31
      const unsigned short* aBase = sA + bufOff;
      const unsigned short* bBase = sB + bufOff;
      FragCast a[2], b[4];
      a[0].q[0] = *(const uint4*)&aBase[aOff0];
      a[0].q[1] = *(const uint4*)&aBase[aOff0 + 16];
      a[1].q[0] = *(const uint4*)&aBase[aOff1];
      a[1].q[1] = *(const uint4*)&aBase[aOff1 + 16];
      #pragma unroll
      for (int tn = 0; tn < 4; ++tn) {
        b[tn].q[0] = *(const uint4*)&bBase[bOff + tn * (16 * 40)];
        b[tn].q[1] = *(const uint4*)&bBase[bOff + tn * (16 * 40) + 16];
      }

      #pragma unroll
      for (int tm = 0; tm < 2; ++tm)
        #pragma unroll
        for (int tn = 0; tn < 4; ++tn)
          c[tm][tn] = __builtin_amdgcn_wmma_f32_16x16x32_bf16(
              false, a[tm].v, false, b[tn].v, (short)0, c[tm][tn], false, false);

      // stage next tile into the other buffer (loads already in flight)
      if (step + 1 < KSTEPS) {
        const int nxt = bufOff ^ ABUF;
        *(uint4*)&sA[nxt + sOff0] = ha0;  *(uint4*)&sA[nxt + sOff1] = ha1;
        *(uint4*)&sB[nxt + sOff0] = wv0;  *(uint4*)&sB[nxt + sOff1] = wv1;
      }
      bufOff ^= ABUF;
    }

    // ---- spill logits tile to LDS (C layout: VGPR r -> M=r(+8), N=lane&15) ----
    #pragma unroll
    for (int tm = 0; tm < 2; ++tm)
      #pragma unroll
      for (int tn = 0; tn < 4; ++tn)
        #pragma unroll
        for (int r = 0; r < 8; ++r) {
          const int row = wm * 32 + tm * 16 + r + laneHalf * 8;
          const int col = wn * 64 + tn * 16 + ln;
          sLogits[row * 132 + col] = c[tm][tn][r];
        }
    __syncthreads();

    // ---- per-token chunk max / sumexp (two threads per token, float4 reads) ----
    const int t = tid & 127, half = tid >> 7;
    const float4* rp = (const float4*)&sLogits[t * 132 + half * 64];
    float m = -INFINITY;
    #pragma unroll
    for (int j = 0; j < 16; ++j) {
      const float4 v = rp[j];
      m = fmaxf(m, fmaxf(fmaxf(v.x, v.y), fmaxf(v.z, v.w)));
    }
    sRed[half * 128 + t] = m;
    __syncthreads();
    const float cm = fmaxf(sRed[t], sRed[128 + t]);
    __syncthreads();
    float s = 0.0f;
    #pragma unroll
    for (int j = 0; j < 16; ++j) {
      const float4 v = rp[j];
      s += __expf(v.x - cm) + __expf(v.y - cm) + __expf(v.z - cm) + __expf(v.w - cm);
    }
    sRed[half * 128 + t] = s;
    __syncthreads();

    if (tid < 128) {
      const float s2 = sRed[t] + sRed[128 + t];
      const float om = sMax[t];
      const float nm = fmaxf(om, cm);
      sSum[t] = sSum[t] * __expf(om - nm) + s2 * __expf(cm - nm);
      sMax[t] = nm;
      if (myTgt >= col0 && myTgt < col0 + TN)    // unique writer over the grid
        pTgt[token0 + t] = sLogits[t * 132 + (myTgt - col0)];
    }
    __syncthreads();
  }

  if (tid < 128) {
    pMax[blockIdx.y * N_TOK + token0 + tid] = sMax[tid];
    pSum[blockIdx.y * N_TOK + token0 + tid] = sSum[tid];
  }
}

// ---------------------------------------------------------------------------
// Kernel 2: combine stripe partials per token -> loss per token (deterministic)
// ---------------------------------------------------------------------------
__global__ __launch_bounds__(256) void ce_combine(
    const float* __restrict__ pMax, const float* __restrict__ pSum,
    const float* __restrict__ pTgt, const int* __restrict__ targets,
    float* __restrict__ pLoss, float* __restrict__ pValid)
{
  const int t = blockIdx.x * 256 + threadIdx.x;
  if (t >= N_TOK) return;
  float M = -INFINITY;
  for (int c = 0; c < NCHUNKS; ++c) M = fmaxf(M, pMax[c * N_TOK + t]);
  float S = 0.0f;
  for (int c = 0; c < NCHUNKS; ++c)
    S += pSum[c * N_TOK + t] * __expf(pMax[c * N_TOK + t] - M);
  const bool valid = (targets[t] != IGNORE_INDEX);
  const float loss = (M + logf(S)) - pTgt[t];
  pLoss[t]  = valid ? loss : 0.0f;
  pValid[t] = valid ? 1.0f : 0.0f;
}

// ---------------------------------------------------------------------------
// Kernel 3: single-block deterministic tree reduction -> scalar mean loss
// ---------------------------------------------------------------------------
__global__ __launch_bounds__(256) void ce_reduce(
    const float* __restrict__ pLoss, const float* __restrict__ pValid,
    float* __restrict__ out)
{
  __shared__ float sl[256], sv[256];
  float a = 0.0f, b = 0.0f;
  for (int i = threadIdx.x; i < N_TOK; i += 256) { a += pLoss[i]; b += pValid[i]; }
  sl[threadIdx.x] = a; sv[threadIdx.x] = b;
  __syncthreads();
  for (int s = 128; s > 0; s >>= 1) {
    if (threadIdx.x < s) {
      sl[threadIdx.x] += sl[threadIdx.x + s];
      sv[threadIdx.x] += sv[threadIdx.x + s];
    }
    __syncthreads();
  }
  if (threadIdx.x == 0) out[0] = sl[0] / fmaxf(sv[0], 1.0f);
}

// ---------------------------------------------------------------------------
extern "C" void kernel_launch(void* const* d_in, const int* in_sizes, int n_in,
                              void* d_out, int out_size, void* d_ws, size_t ws_size,
                              hipStream_t stream) {
  const float* H       = (const float*)d_in[0];
  const float* W       = (const float*)d_in[1];
  const int*   targets = (const int*)d_in[2];

  // workspace layout: float partials first, then bf16 copies of H and W.
  // total: ~2.2 MB + 16 MB + 256 MB  (assumes d_ws sized generously)
  float* ws     = (float*)d_ws;
  float* pMax   = ws;                                 // [NCHUNKS][N_TOK]
  float* pSum   = pMax + (size_t)NCHUNKS * N_TOK;     // [NCHUNKS][N_TOK]
  float* pTgt   = pSum + (size_t)NCHUNKS * N_TOK;     // [N_TOK]
  float* pLoss  = pTgt + N_TOK;                       // [N_TOK]
  float* pValid = pLoss + N_TOK;                      // [N_TOK]
  unsigned short* Hb = (unsigned short*)(pValid + N_TOK);          // 16B-aligned
  unsigned short* Wb = Hb + (size_t)N_TOK * DMODEL;
  (void)in_sizes; (void)n_in; (void)out_size; (void)ws_size;

  cvt_bf16<<<1024, 256, 0, stream>>>(H, Hb, N_TOK * (DMODEL / 4));
  cvt_bf16<<<8192, 256, 0, stream>>>(W, Wb, VOCAB * (DMODEL / 4));

  dim3 grid(N_TOK / TM, NCHUNKS);   // x-fastest: token tiles share each W stripe in L2
  ce_main<<<grid, 256, 0, stream>>>(Hb, Wb, targets, pMax, pSum, pTgt);
  ce_combine<<<N_TOK / 256, 256, 0, stream>>>(pMax, pSum, pTgt, targets, pLoss, pValid);
  ce_reduce<<<1, 256, 0, stream>>>(pLoss, pValid, (float*)d_out);
}
